// MoeBlock_58703613002488
// MI455X (gfx1250) — matrix operand order repaired and checked by reading, not compiled, not run on hardware
//
#include <hip/hip_runtime.h>
#include <hip/hip_bf16.h>

typedef _Float16 f16;
typedef __attribute__((ext_vector_type(16))) _Float16 v16h;
typedef __attribute__((ext_vector_type(8)))  float    v8f;
typedef __attribute__((ext_vector_type(4)))  unsigned int u32x4;

#define WMMA_F16(A,B,C) __builtin_amdgcn_wmma_f32_16x16x32_f16(false,(A),false,(B),(short)0,(C),false,false)

// async global->LDS copy of 16 bytes per lane (ASYNCcnt-tracked)
#define ASYNC_CP_B128(ldsaddr32, gaddr64)                                   \
    asm volatile("global_load_async_to_lds_b128 %0, %1, off"                \
                 :: "v"(ldsaddr32), "v"(gaddr64) : "memory")

static __device__ __forceinline__ v8f vzero8() {
    v8f z;
#pragma unroll
    for (int i = 0; i < 8; i++) z[i] = 0.f;
    return z;
}
static __device__ __forceinline__ v16h hzero16() {
    v16h z;
#pragma unroll
    for (int i = 0; i < 16; i++) z[i] = (_Float16)0;
    return z;
}

// A-fragment (16x32 f16): lane half h; K=[16h..16h+15] contiguous
static __device__ __forceinline__ v16h load_a(const f16* p, int h) {
    union { u32x4 u[2]; v16h v; } x;
    x.u[0] = *(const u32x4*)(p + 16 * h);
    x.u[1] = *(const u32x4*)(p + 16 * h + 8);
    return x.v;
}

// ============ shared WMMA core: 32(M)x64(N) per wave, 256x64 per block =======
// B tile (64 x 32 f16 per k-step) staged into double-buffered LDS by async
// copies: thread t copies 16B; s_wait_asynccnt + barrier; B-frags read from LDS.
// Kd is a runtime value so the k-loop stays a compact resident loop (no full
// unroll -> no I-cache blowout); steady state is branch-free, last step peeled.
static __device__ __forceinline__ void wmma_core(
    const f16* __restrict__ a0, bool v0,     // A row ptr for rows m0+q
    const f16* __restrict__ a1, bool v1,     // A row ptr for rows m0+16+q
    const f16* __restrict__ BtN,             // Bt + n0*Kd  (f16 [N][Kd])
    int Kd,
    f16* bt,                                 // LDS: 2 stages x 64*32 halves
    v8f acc[8]) {
    const int tid = threadIdx.x;
    const int lane = tid & 31, h = lane >> 4, q = lane & 15;
    const int sn = tid >> 2, sc = tid & 3;   // staging: row 0..63, 16B chunk 0..3
    const f16* gsrc = BtN + (long long)sn * Kd + sc * 8;
    unsigned lds0 = (unsigned)(unsigned long long)bt + (unsigned)(sn * 32 + sc * 8) * 2;
    const v16h az = hzero16();
    const int ns = Kd >> 5;

    ASYNC_CP_B128(lds0, (unsigned long long)gsrc);           // stage 0

    auto compute = [&](int st, int k0) {
        const f16* bts = bt + st * 2048;
        v16h af0 = v0 ? load_a(a0 + k0, h) : az;
        v16h af1 = v1 ? load_a(a1 + k0, h) : az;
        v16h b0 = *(const v16h*)(bts + (q) * 32 + 16 * h);
        v16h b1 = *(const v16h*)(bts + (16 + q) * 32 + 16 * h);
        v16h b2 = *(const v16h*)(bts + (32 + q) * 32 + 16 * h);
        v16h b3 = *(const v16h*)(bts + (48 + q) * 32 + 16 * h);
        acc[0] = WMMA_F16(af0, b0, acc[0]);
        acc[4] = WMMA_F16(af1, b0, acc[4]);
        acc[1] = WMMA_F16(af0, b1, acc[1]);
        acc[5] = WMMA_F16(af1, b1, acc[5]);
        acc[2] = WMMA_F16(af0, b2, acc[2]);
        acc[6] = WMMA_F16(af1, b2, acc[6]);
        acc[3] = WMMA_F16(af0, b3, acc[3]);
        acc[7] = WMMA_F16(af1, b3, acc[7]);
    };

#pragma unroll 2
    for (int i = 0; i < ns - 1; i++) {
        ASYNC_CP_B128(lds0 + (unsigned)(((i & 1) ^ 1) * 4096),
                      (unsigned long long)(gsrc + (i + 1) * 32));
        asm volatile("s_wait_asynccnt 0x1" ::: "memory");
        __syncthreads();
        compute(i & 1, i * 32);
        __syncthreads();
    }
    asm volatile("s_wait_asynccnt 0x0" ::: "memory");
    __syncthreads();
    compute((ns - 1) & 1, (ns - 1) * 32);
}

// ---------------- weight prep: fp32 [K x N] (row stride rs) -> f16 [N x K] ----
__global__ void k_transpose_cvt(const float* __restrict__ src, int rs,
                                f16* __restrict__ dst, int Krows, int Ncols) {
    long long idx = (long long)blockIdx.x * 256 + threadIdx.x;
    long long total = (long long)Krows * Ncols;
    if (idx >= total) return;
    int k = (int)(idx / Ncols), n = (int)(idx % Ncols);
    dst[(long long)n * Krows + k] = (f16)src[(long long)k * rs + n];
}

// ---------------- layernorm (C=1024) -> f16 ----------------------------------
__global__ void k_ln_f16(const float* __restrict__ x, const float* __restrict__ w,
                         f16* __restrict__ out) {
    int row = blockIdx.x, tid = threadIdx.x;
    const float4 v = ((const float4*)(x + (long long)row * 1024))[tid];
    float s  = v.x + v.y + v.z + v.w;
    float ss = v.x*v.x + v.y*v.y + v.z*v.z + v.w*v.w;
    __shared__ float sh[256], sh2[256];
    sh[tid] = s; sh2[tid] = ss;
    __syncthreads();
    for (int o = 128; o > 0; o >>= 1) {
        if (tid < o) { sh[tid] += sh[tid + o]; sh2[tid] += sh2[tid + o]; }
        __syncthreads();
    }
    float mean = sh[0] * (1.f / 1024.f);
    float var  = sh2[0] * (1.f / 1024.f) - mean * mean;
    float inv  = rsqrtf(var + 1e-5f);
    const float4 wv = ((const float4*)w)[tid];
    f16* o4 = out + (long long)row * 1024 + tid * 4;
    o4[0] = (f16)((v.x - mean) * inv * wv.x);
    o4[1] = (f16)((v.y - mean) * inv * wv.y);
    o4[2] = (f16)((v.z - mean) * inv * wv.z);
    o4[3] = (f16)((v.w - mean) * inv * wv.w);
}

// ---------------- GEMM: f16 out ----------------------------------------------
__global__ __launch_bounds__(256) void k_gemm_f16(const f16* __restrict__ A,
                                                  const f16* __restrict__ Bt,
                                                  f16* __restrict__ C,
                                                  int N, int Kd) {
    __shared__ __align__(64) f16 bt[2 * 2048];
    int lane = threadIdx.x & 31, wv = threadIdx.x >> 5;
    int h = lane >> 4, q = lane & 15;
    int m0 = blockIdx.y * 256 + wv * 32;
    int n0 = blockIdx.x * 64;
    v8f acc[8];
#pragma unroll
    for (int i = 0; i < 8; i++) acc[i] = vzero8();
    wmma_core(A + (long long)(m0 + q) * Kd, true,
              A + (long long)(m0 + 16 + q) * Kd, true,
              Bt + (long long)n0 * Kd, Kd, bt, acc);
#pragma unroll
    for (int s = 0; s < 2; s++)
#pragma unroll
        for (int t = 0; t < 4; t++) {
            int col = n0 + 16 * t + q;
#pragma unroll
            for (int r = 0; r < 8; r++)
                C[(long long)(m0 + s * 16 + h * 8 + r) * N + col] = (f16)acc[s * 4 + t][r];
        }
}

// ---------------- GEMM: f32 out + residual -----------------------------------
__global__ __launch_bounds__(256) void k_gemm_f32_res(const f16* __restrict__ A,
                                                      const f16* __restrict__ Bt,
                                                      float* __restrict__ C,
                                                      const float* __restrict__ res,
                                                      int N, int Kd) {
    __shared__ __align__(64) f16 bt[2 * 2048];
    int lane = threadIdx.x & 31, wv = threadIdx.x >> 5;
    int h = lane >> 4, q = lane & 15;
    int m0 = blockIdx.y * 256 + wv * 32;
    int n0 = blockIdx.x * 64;
    v8f acc[8];
#pragma unroll
    for (int i = 0; i < 8; i++) acc[i] = vzero8();
    wmma_core(A + (long long)(m0 + q) * Kd, true,
              A + (long long)(m0 + 16 + q) * Kd, true,
              Bt + (long long)n0 * Kd, Kd, bt, acc);
#pragma unroll
    for (int s = 0; s < 2; s++)
#pragma unroll
        for (int t = 0; t < 4; t++) {
            int col = n0 + 16 * t + q;
#pragma unroll
            for (int r = 0; r < 8; r++) {
                long long o = (long long)(m0 + s * 16 + h * 8 + r) * N + col;
                C[o] = acc[s * 4 + t][r] + res[o];
            }
        }
}

// ---------------- qkv split: [4096 x 3072] f16 -> Q(scaled)/K/V^T -------------
__global__ void k_qkv_split(const f16* __restrict__ qkv, f16* __restrict__ Q,
                            f16* __restrict__ K, f16* __restrict__ VT) {
    long long idx = (long long)blockIdx.x * 256 + threadIdx.x;  // total 4096*3072
    int n = (int)(idx / 3072), j = (int)(idx % 3072);
    int which = j >> 10, c = j & 1023;
    int hh = c >> 6, d = c & 63;
    int b = n >> 11, t = n & 2047;
    f16 v = qkv[idx];
    long long hb = (long long)(b * 16 + hh);
    if (which == 0)      Q[(hb * 2048 + t) * 64 + d] = (f16)((float)v * 0.125f);
    else if (which == 1) K[(hb * 2048 + t) * 64 + d] = v;
    else                 VT[(hb * 64 + d) * 2048 + t] = v;
}

// ---------------- flash attention: 1 wave per 16 queries ----------------------
__global__ __launch_bounds__(32) void k_attn(const f16* __restrict__ Q,
                                             const f16* __restrict__ Kf,
                                             const f16* __restrict__ VT,
                                             f16* __restrict__ Y) {
    const int Tl = 2048;
    int lane = threadIdx.x;
    int h = lane >> 4, q = lane & 15;
    int q0 = blockIdx.x * 16;
    int bh = blockIdx.y, bb = bh >> 4, hh = bh & 15;
    const f16* Qh = Q  + (long long)bh * Tl * 64;
    const f16* Kh = Kf + (long long)bh * Tl * 64;
    const f16* Vh = VT + (long long)bh * 64 * Tl;

    v16h qa0 = load_a(Qh + (long long)(q0 + q) * 64, h);
    v16h qa1 = load_a(Qh + (long long)(q0 + q) * 64 + 32, h);

    v8f o[4];
#pragma unroll
    for (int t = 0; t < 4; t++) o[t] = vzero8();
    float mrow[8], lrow[8];
#pragma unroll
    for (int r = 0; r < 8; r++) { mrow[r] = -INFINITY; lrow[r] = 0.f; }

    __shared__ __align__(64) f16 P[16 * 32];

    int nkb = ((q0 + 15) >> 5) + 1;
    for (int ib = 0; ib < nkb; ib++) {
        int kb = ib * 32;
        v8f s0 = vzero8(), s1 = vzero8();
        {
            int key = kb + q;
            v16h b0 = *(const v16h*)(Kh + (long long)key * 64 + 16 * h);
            v16h b1 = *(const v16h*)(Kh + (long long)key * 64 + 32 + 16 * h);
            s0 = WMMA_F16(qa0, b0, s0);
            s0 = WMMA_F16(qa1, b1, s0);
            key = kb + 16 + q;
            v16h b2 = *(const v16h*)(Kh + (long long)key * 64 + 16 * h);
            v16h b3 = *(const v16h*)(Kh + (long long)key * 64 + 32 + 16 * h);
            s1 = WMMA_F16(qa0, b2, s1);
            s1 = WMMA_F16(qa1, b3, s1);
        }
        float p0[8], p1[8], alpha[8];
        int col0 = kb + q, col1 = kb + 16 + q;
#pragma unroll
        for (int r = 0; r < 8; r++) {
            int qr = q0 + h * 8 + r;
            float v0 = (col0 <= qr) ? s0[r] : -INFINITY;
            float v1 = (col1 <= qr) ? s1[r] : -INFINITY;
            float mx = fmaxf(v0, v1);
            mx = fmaxf(mx, __shfl_xor(mx, 1, 32));
            mx = fmaxf(mx, __shfl_xor(mx, 2, 32));
            mx = fmaxf(mx, __shfl_xor(mx, 4, 32));
            mx = fmaxf(mx, __shfl_xor(mx, 8, 32));
            float mn = fmaxf(mrow[r], mx);
            p0[r] = expf(v0 - mn);
            p1[r] = expf(v1 - mn);
            float rs = p0[r] + p1[r];
            rs += __shfl_xor(rs, 1, 32);
            rs += __shfl_xor(rs, 2, 32);
            rs += __shfl_xor(rs, 4, 32);
            rs += __shfl_xor(rs, 8, 32);
            alpha[r] = expf(mrow[r] - mn);
            lrow[r] = lrow[r] * alpha[r] + rs;
            mrow[r] = mn;
        }
#pragma unroll
        for (int t = 0; t < 4; t++)
#pragma unroll
            for (int r = 0; r < 8; r++) o[t][r] *= alpha[r];
#pragma unroll
        for (int r = 0; r < 8; r++) {
            P[(h * 8 + r) * 32 + q]      = (f16)p0[r];
            P[(h * 8 + r) * 32 + 16 + q] = (f16)p1[r];
        }
        asm volatile("s_wait_dscnt 0x0" ::: "memory");
        v16h pa = load_a((const f16*)P + q * 32, h);
#pragma unroll
        for (int t = 0; t < 4; t++) {
            v16h vb = *(const v16h*)(Vh + (long long)(16 * t + q) * Tl + kb + 16 * h);
            o[t] = WMMA_F16(pa, vb, o[t]);
        }
    }
#pragma unroll
    for (int t = 0; t < 4; t++)
#pragma unroll
        for (int r = 0; r < 8; r++) {
            int qr = q0 + h * 8 + r;
            Y[(long long)(bb * 2048 + qr) * 1024 + hh * 64 + 16 * t + q]
                = (f16)(o[t][r] / lrow[r]);
        }
}

// ---------------- router: logits + softmax + top-2 ----------------------------
__global__ void k_router(const f16* __restrict__ xn2, const float* __restrict__ rwgt,
                         float* __restrict__ rw_out, int* __restrict__ sel_out) {
    int tok = blockIdx.x, tid = threadIdx.x;
    int e = tid >> 5, lane = tid & 31;
    const f16* xr = xn2 + (long long)tok * 1024;
    float p = 0.f;
    for (int i = 0; i < 32; i++) {
        int c = lane + i * 32;
        p += (float)xr[c] * rwgt[c * 8 + e];
    }
    for (int m = 16; m > 0; m >>= 1) p += __shfl_xor(p, m, 32);
    __shared__ float lg[8];
    if (lane == 0) lg[e] = p;
    __syncthreads();
    if (tid == 0) {
        float mx = lg[0];
        for (int i = 1; i < 8; i++) mx = fmaxf(mx, lg[i]);
        float ex[8];
        for (int i = 0; i < 8; i++) ex[i] = expf(lg[i] - mx);
        int i0 = 0;
        for (int i = 1; i < 8; i++) if (ex[i] > ex[i0]) i0 = i;
        int i1 = (i0 == 0) ? 1 : 0;
        for (int i = 0; i < 8; i++) if (i != i0 && ex[i] > ex[i1]) i1 = i;
        float s2 = ex[i0] + ex[i1];
        sel_out[tok * 2] = i0; sel_out[tok * 2 + 1] = i1;
        rw_out[tok * 2] = ex[i0] / s2; rw_out[tok * 2 + 1] = ex[i1] / s2;
    }
}

// ---------------- MoE bookkeeping (expert segments padded to 256 rows) --------
__global__ void k_zero_counts(int* counts) { if (threadIdx.x < 8) counts[threadIdx.x] = 0; }

__global__ void k_fill_rows(int* gather_tok, float* row_rw) {
    int i = blockIdx.x * 256 + threadIdx.x;
    if (i < 10240) { gather_tok[i] = -1; row_rw[i] = 0.f; }
}

__global__ void k_assign(const int* sel, int* counts, int* posbuf) {
    int g = blockIdx.x * 256 + threadIdx.x;
    if (g < 8192) posbuf[g] = atomicAdd(&counts[sel[g]], 1);
}

__global__ void k_offsets(const int* counts, int* poff) {
    if (threadIdx.x == 0 && blockIdx.x == 0) {
        int run = 0;
        for (int e = 0; e < 8; e++) { poff[e] = run; run += ((counts[e] + 255) / 256) * 256; }
        poff[8] = run;
    }
}

__global__ void k_scatter(const int* sel, const float* rw, const int* posbuf,
                          const int* poff, int* gather_tok, float* row_rw, int* row_of) {
    int g = blockIdx.x * 256 + threadIdx.x;
    if (g < 8192) {
        int row = poff[sel[g]] + posbuf[g];
        gather_tok[row] = g >> 1;
        row_rw[row] = rw[g];
        row_of[g] = row;
    }
}

// ---------------- MoE GEMM1: h = gelu(gather(xn2) @ w1_e) * rw ----------------
__global__ __launch_bounds__(256) void k_moe_gemm1(const f16* __restrict__ X,
                                                   const f16* __restrict__ w1t,
                                                   f16* __restrict__ H,
                                                   const int* __restrict__ gather_tok,
                                                   const float* __restrict__ row_rw,
                                                   const int* __restrict__ poff,
                                                   int Kd, int F) {
    __shared__ __align__(64) f16 bt[2 * 2048];
    int lane = threadIdx.x & 31, wv = threadIdx.x >> 5;
    int h = lane >> 4, q = lane & 15;
    int m0 = blockIdx.y * 256 + wv * 32;
    int n0 = blockIdx.x * 64;
    int tileStart = blockIdx.y * 256;
    int e = 0;
    for (int i = 1; i < 8; i++) if (tileStart >= poff[i]) e = i;
    int tok0 = gather_tok[m0 + q];
    int tok1 = gather_tok[m0 + 16 + q];
    v8f acc[8];
#pragma unroll
    for (int i = 0; i < 8; i++) acc[i] = vzero8();
    wmma_core(X + (long long)((tok0 >= 0) ? tok0 : 0) * Kd, tok0 >= 0,
              X + (long long)((tok1 >= 0) ? tok1 : 0) * Kd, tok1 >= 0,
              w1t + ((long long)e * F + n0) * Kd, Kd, bt, acc);
#pragma unroll
    for (int s = 0; s < 2; s++)
#pragma unroll
        for (int t = 0; t < 4; t++) {
            int col = n0 + 16 * t + q;
#pragma unroll
            for (int r = 0; r < 8; r++) {
                int row = m0 + s * 16 + h * 8 + r;
                int st = gather_tok[row];
                float w = (st >= 0) ? row_rw[row] : 0.f;
                float x = acc[s * 4 + t][r];
                float g = 0.5f * x * (1.f + erff(x * 0.70710678f));
                H[(long long)row * F + col] = (f16)(g * w);
            }
        }
}

// ---------------- MoE GEMM2: y_rows = h @ w2_e --------------------------------
__global__ __launch_bounds__(256) void k_moe_gemm2(const f16* __restrict__ H,
                                                   const f16* __restrict__ w2t,
                                                   float* __restrict__ Yr,
                                                   const int* __restrict__ poff,
                                                   int Kd, int N) {
    __shared__ __align__(64) f16 bt[2 * 2048];
    int lane = threadIdx.x & 31, wv = threadIdx.x >> 5;
    int h = lane >> 4, q = lane & 15;
    int m0 = blockIdx.y * 256 + wv * 32;
    int n0 = blockIdx.x * 64;
    int tileStart = blockIdx.y * 256;
    int e = 0;
    for (int i = 1; i < 8; i++) if (tileStart >= poff[i]) e = i;
    v8f acc[8];
#pragma unroll
    for (int i = 0; i < 8; i++) acc[i] = vzero8();
    wmma_core(H + (long long)(m0 + q) * Kd, true,
              H + (long long)(m0 + 16 + q) * Kd, true,
              w2t + ((long long)e * N + n0) * Kd, Kd, bt, acc);
#pragma unroll
    for (int s = 0; s < 2; s++)
#pragma unroll
        for (int t = 0; t < 4; t++) {
            int col = n0 + 16 * t + q;
#pragma unroll
            for (int r = 0; r < 8; r++)
                Yr[(long long)(m0 + s * 16 + h * 8 + r) * N + col] = acc[s * 4 + t][r];
        }
}

// ---------------- final combine ----------------------------------------------
__global__ void k_final(const float* __restrict__ x1, const float* __restrict__ yr,
                        const int* __restrict__ row_of, float* __restrict__ out) {
    int tok = blockIdx.x, tid = threadIdx.x;
    int g0 = row_of[tok * 2], g1 = row_of[tok * 2 + 1];
    const float4 a = ((const float4*)(x1 + (long long)tok * 1024))[tid];
    const float4 b = ((const float4*)(yr + (long long)g0 * 1024))[tid];
    const float4 c = ((const float4*)(yr + (long long)g1 * 1024))[tid];
    float4 o;
    o.x = a.x + b.x + c.x; o.y = a.y + b.y + c.y;
    o.z = a.z + b.z + c.z; o.w = a.w + b.w + c.w;
    ((float4*)(out + (long long)tok * 1024))[tid] = o;
}

extern "C" void kernel_launch(void* const* d_in, const int* in_sizes, int n_in,
                              void* d_out, int out_size, void* d_ws, size_t ws_size,
                              hipStream_t stream) {
    (void)in_sizes; (void)n_in; (void)out_size; (void)ws_size;
    const float* x        = (const float*)d_in[0];
    const float* ln1_w    = (const float*)d_in[1];
    const float* c_attn_w = (const float*)d_in[2];
    const float* c_proj_w = (const float*)d_in[3];
    const float* ln2_w    = (const float*)d_in[4];
    const float* router_w = (const float*)d_in[5];
    const float* w1       = (const float*)d_in[6];
    const float* w2       = (const float*)d_in[7];
    float* out = (float*)d_out;

    const long long NTOK = 4096, Cd = 1024, NC3 = 3072, Fd = 2048, TR = 10240;

    size_t off = 0;
    auto carve = [&](size_t bytes) -> char* {
        char* p = (char*)d_ws + off;
        off = (off + bytes + 255) & ~(size_t)255;
        return p;
    };
    f16*   wAttnT = (f16*)  carve(NC3 * Cd * 2);
    f16*   wProjT = (f16*)  carve(Cd * Cd * 2);
    f16*   w1t    = (f16*)  carve(8 * Fd * Cd * 2);
    f16*   w2t    = (f16*)  carve(8 * Cd * Fd * 2);
    f16*   xn1    = (f16*)  carve(NTOK * Cd * 2);
    f16*   qkv    = (f16*)  carve(NTOK * NC3 * 2);
    f16*   Qb     = (f16*)  carve(NTOK * Cd * 2);
    f16*   Kb     = (f16*)  carve(NTOK * Cd * 2);
    f16*   VTb    = (f16*)  carve(NTOK * Cd * 2);
    f16*   yatt   = (f16*)  carve(NTOK * Cd * 2);
    float* x1     = (float*)carve(NTOK * Cd * 4);
    f16*   xn2    = (f16*)  carve(NTOK * Cd * 2);
    float* rw     = (float*)carve(8192 * 4);
    int*   sel    = (int*)  carve(8192 * 4);
    int*   counts = (int*)  carve(8 * 4);
    int*   poff   = (int*)  carve(9 * 4);
    int*   posbuf = (int*)  carve(8192 * 4);
    int*   row_of = (int*)  carve(8192 * 4);
    int*   gtok   = (int*)  carve(TR * 4);
    float* row_rw = (float*)carve(TR * 4);
    f16*   Hbuf   = (f16*)  carve(TR * Fd * 2);
    float* yrows  = (float*)carve(TR * Cd * 4);

    // weight prep
    k_transpose_cvt<<<(1024 * 3072 + 255) / 256, 256, 0, stream>>>(c_attn_w, 3072, wAttnT, 1024, 3072);
    k_transpose_cvt<<<(1024 * 1024 + 255) / 256, 256, 0, stream>>>(c_proj_w, 1024, wProjT, 1024, 1024);
    for (int e = 0; e < 8; e++) {
        k_transpose_cvt<<<(1024 * 2048 + 255) / 256, 256, 0, stream>>>(
            w1 + (long long)e * 2048, 16384, w1t + (long long)e * 2048 * 1024, 1024, 2048);
        k_transpose_cvt<<<(2048 * 1024 + 255) / 256, 256, 0, stream>>>(
            w2 + (long long)e * 2048 * 1024, 1024, w2t + (long long)e * 1024 * 2048, 2048, 1024);
    }

    // attention path
    k_ln_f16<<<4096, 256, 0, stream>>>(x, ln1_w, xn1);
    k_gemm_f16<<<dim3(48, 16), 256, 0, stream>>>(xn1, wAttnT, qkv, 3072, 1024);
    k_qkv_split<<<(int)((NTOK * NC3 + 255) / 256), 256, 0, stream>>>(qkv, Qb, Kb, VTb);
    k_attn<<<dim3(128, 32), 32, 0, stream>>>(Qb, Kb, VTb, yatt);
    k_gemm_f32_res<<<dim3(16, 16), 256, 0, stream>>>(yatt, wProjT, x1, x, 1024, 1024);

    // MoE path
    k_ln_f16<<<4096, 256, 0, stream>>>(x1, ln2_w, xn2);
    k_router<<<4096, 256, 0, stream>>>(xn2, router_w, rw, sel);
    k_zero_counts<<<1, 32, 0, stream>>>(counts);
    k_fill_rows<<<40, 256, 0, stream>>>(gtok, row_rw);
    k_assign<<<32, 256, 0, stream>>>(sel, counts, posbuf);
    k_offsets<<<1, 32, 0, stream>>>(counts, poff);
    k_scatter<<<32, 256, 0, stream>>>(sel, rw, posbuf, poff, gtok, row_rw, row_of);
    k_moe_gemm1<<<dim3(32, 40), 256, 0, stream>>>(xn2, w1t, Hbuf, gtok, row_rw, poff, 1024, 2048);
    k_moe_gemm2<<<dim3(16, 40), 256, 0, stream>>>(Hbuf, w2t, yrows, poff, 2048, 1024);
    k_final<<<4096, 256, 0, stream>>>(x1, yrows, row_of, out);
}
